// MaxGraphWhisperAttention_81114752352871
// MI455X (gfx1250) — matrix operand order repaired and checked by reading, not compiled, not run on hardware
//
#include <hip/hip_runtime.h>

// ---------------- problem constants ----------------
#define E_DIM   1280
#define NHEAD   20
#define HDIM    64
#define SEQ     1500
#define SEQP    1504          // padded to multiple of 32
#define BATCH   4
#define MROWS   6000          // BATCH*SEQ  (exactly 375 * 16)
#define QTILES  94            // ceil(1500/16)
#define BLOCKS_PER_BH 24      // 24 blocks * 4 waves = 96 >= 94 q-tiles
#define GEMM_TILES (375 * 20) // 16x64 wave tiles
#define KT_FULL 1472          // first partial 32-col tile start (masked tail)

typedef __attribute__((ext_vector_type(16))) __bf16 v16bf;
typedef __attribute__((ext_vector_type(8)))  float  v8f;

union Frag32B { uint4 q[2]; v16bf v; };
union FragPK  { unsigned u[8]; v16bf v; };

__device__ __forceinline__ unsigned short f2bf(float f) {
  unsigned u = __float_as_uint(f);
  u += 0x7FFFu + ((u >> 16) & 1u);          // round-to-nearest-even
  return (unsigned short)(u >> 16);
}

__device__ __forceinline__ v8f v8f_zero() {
  v8f z = {0.f, 0.f, 0.f, 0.f, 0.f, 0.f, 0.f, 0.f};
  return z;
}

// Wave-local LDS ordering: hardware keeps a wave's LDS ops in order; we only
// need to stop the compiler from reordering the cross-lane store->load.
__device__ __forceinline__ void wave_lds_sync() {
  __builtin_amdgcn_fence(__ATOMIC_ACQ_REL, "wavefront");
  __builtin_amdgcn_wave_barrier();
}

// ---------------- elementwise helpers ----------------
__global__ void f32_to_bf16_kernel(const float* __restrict__ in,
                                   unsigned short* __restrict__ out, int n) {
  int i = blockIdx.x * blockDim.x + threadIdx.x;
  if (i < n) out[i] = f2bf(in[i]);
}

__global__ void zero_u16_kernel(unsigned short* __restrict__ p, int n) {
  int stride = gridDim.x * blockDim.x;
  for (int i = blockIdx.x * blockDim.x + threadIdx.x; i < n; i += stride) p[i] = 0;
}

// ---------------- generic WMMA GEMM:  C = A[M,K] * W[N,K]^T (+bias) --------
// One wave computes a 16(M) x 64(N) tile: 1 A-frag reused by 4 WMMAs per k-step.
// MODE 0: Q  -> (acc+bias)/64 -> bf16 at [b,h,t,d]   ([BH][SEQP][HDIM])
// MODE 1: K  ->  acc          -> bf16 at [b,h,t,d]
// MODE 2: V  ->  acc+bias     -> bf16 at [b,h,d,t]   (transposed, [BH][HDIM][SEQP])
// MODE 3: O  ->  acc+bias     -> f32  at [M,N] row-major (d_out)
template <int MODE>
__global__ __launch_bounds__(256) void
gemm_wmma_kernel(const unsigned short* __restrict__ A,
                 const unsigned short* __restrict__ W,
                 const float* __restrict__ bias,
                 void* __restrict__ out) {
  const int K = E_DIM, N = E_DIM;
  const int tilesN64 = N / 64;                // 20
  int wv   = threadIdx.x >> 5;
  int lane = threadIdx.x & 31;
  int hl   = lane >> 4;                       // lane half (wave32)
  int ln   = lane & 15;
  int tile = blockIdx.x * 8 + wv;
  if (tile >= GEMM_TILES) return;
  int tm = tile / tilesN64;
  int tn = tile % tilesN64;

  const unsigned short* Arow = A + (size_t)(tm * 16 + ln) * K;  // A: lane = row M
  const unsigned short* Wrow = W + (size_t)(tn * 64 + ln) * K;  // B: lane = col N

  v8f acc[4];
#pragma unroll
  for (int c = 0; c < 4; ++c) acc[c] = v8f_zero();

#pragma unroll 2
  for (int kb = 0; kb < K; kb += 32) {
    int kp = (kb + 32 < K) ? kb + 32 : 0;     // clamped: stay inside the row
    __builtin_prefetch(Arow + kp, 0, 1);      // -> global_prefetch_b8
    Frag32B a;
    // A frag: K = (e%8) + (e/8)*16 + hl*8  -> two contiguous 16B chunks
    a.q[0] = *(const uint4*)(Arow + kb + hl * 8);
    a.q[1] = *(const uint4*)(Arow + kb + 16 + hl * 8);
#pragma unroll
    for (int c = 0; c < 4; ++c) {
      const unsigned short* wr = Wrow + (size_t)c * 16 * K + kb;
      Frag32B b;
      // B frag: K = (e%8) + (e/8)*16 + hl*8 -> two contiguous 16B chunks
      b.q[0] = *(const uint4*)(wr + hl * 8);
      b.q[1] = *(const uint4*)(wr + 16 + hl * 8);
      acc[c] = __builtin_amdgcn_wmma_f32_16x16x32_bf16(false, a.v, false, b.v,
                                                       (short)0, acc[c], false, false);
    }
  }

#pragma unroll
  for (int c = 0; c < 4; ++c) {
    int gn = tn * 64 + c * 16 + ln;
    float bv = (MODE == 1) ? 0.f : bias[gn];
#pragma unroll
    for (int r = 0; r < 8; ++r) {
      int gm = tm * 16 + r + hl * 8;          // C/D frag: row = r + half*8
      float v = acc[c][r];
      if (MODE == 0)      v = (v + bv) * (1.0f / 64.0f);  // both sqrt(D) scalings
      else if (MODE != 1) v += bv;
      if (MODE == 3) {
        ((float*)out)[(size_t)gm * N + gn] = v;
      } else {
        int b = gm / SEQ, t = gm - b * SEQ;
        int h = gn >> 6, d = gn & 63;
        int bh = b * NHEAD + h;
        unsigned short* o = (unsigned short*)out;
        if (MODE == 2) o[((size_t)bh * HDIM + d) * SEQP + t] = f2bf(v);
        else           o[((size_t)bh * SEQP + t) * HDIM + d] = f2bf(v);
      }
    }
  }
}

// ---------------- flash attention:  softmax(Q K^T) V ----------------
__global__ __launch_bounds__(128) void
attn_flash_kernel(const unsigned short* __restrict__ Q,
                  const unsigned short* __restrict__ Kb,
                  const unsigned short* __restrict__ VT,
                  unsigned short* __restrict__ AO) {
  __shared__ __align__(16) float ssh[4][16][32];   // per-wave raw score tile
  int wv   = threadIdx.x >> 5;
  int lane = threadIdx.x & 31;
  int hl   = lane >> 4;
  int ln   = lane & 15;
  int bh   = blockIdx.x / BLOCKS_PER_BH;
  int qblk = blockIdx.x % BLOCKS_PER_BH;
  int qt   = qblk * 4 + wv;                        // 0..95
  int qtc  = qt < QTILES ? qt : QTILES - 1;        // clamp dummy waves

  // preload Q fragments (16 rows x 64 d) : 2 A-frags (stay resident, no spills)
  const unsigned short* Qrow = Q + ((size_t)bh * SEQP + qtc * 16 + ln) * HDIM;
  Frag32B a0, a1;
  a0.q[0] = *(const uint4*)(Qrow + hl * 8);
  a0.q[1] = *(const uint4*)(Qrow + 16 + hl * 8);
  a1.q[0] = *(const uint4*)(Qrow + 32 + hl * 8);
  a1.q[1] = *(const uint4*)(Qrow + 48 + hl * 8);

  const float NEGINF = -1e30f;
  float mrow = NEGINF;                             // row = ln (replicated in both halves)
  float lrow = 0.f;
  v8f accd[4];
#pragma unroll
  for (int c = 0; c < 4; ++c) accd[c] = v8f_zero();

  // one 32-column k-tile; `masked` is a call-site constant (folded after inline)
  auto kstep = [&](int kt, bool masked) {
    // ---- S = Q K^T for a 16q x 32k tile (two 16-col sub-tiles) ----
    v8f s[2];
    s[0] = v8f_zero(); s[1] = v8f_zero();
#pragma unroll
    for (int j = 0; j < 2; ++j) {
      int tk = kt + j * 16 + ln;                   // lane = key column
      const unsigned short* Krow = Kb + ((size_t)bh * SEQP + tk) * HDIM;
      int tkp = tk + 32 < SEQP ? tk + 32 : tk;     // clamped prefetch of next tile
      __builtin_prefetch(Kb + ((size_t)bh * SEQP + tkp) * HDIM, 0, 1);
      Frag32B b0, b1;
      b0.q[0] = *(const uint4*)(Krow + hl * 8);
      b0.q[1] = *(const uint4*)(Krow + 16 + hl * 8);
      b1.q[0] = *(const uint4*)(Krow + 32 + hl * 8);
      b1.q[1] = *(const uint4*)(Krow + 48 + hl * 8);
      s[j] = __builtin_amdgcn_wmma_f32_16x16x32_bf16(false, a0.v, false, b0.v,
                                                     (short)0, s[j], false, false);
      s[j] = __builtin_amdgcn_wmma_f32_16x16x32_bf16(false, a1.v, false, b1.v,
                                                     (short)0, s[j], false, false);
    }
    // ---- transpose S through LDS (D-layout lane=col -> A-layout lane=row) ---
#pragma unroll
    for (int r = 0; r < 8; ++r) {
      ssh[wv][r + hl * 8][ln]      = s[0][r];
      ssh[wv][r + hl * 8][16 + ln] = s[1][r];
    }
    wave_lds_sync();                               // wave-local: no block barrier
    // Each lane now owns row `ln`; halves split the 32 columns in A-frag order.
    const float4* rowp = (const float4*)&ssh[wv][ln][0];
    float4 q0 = rowp[hl * 2];                      // cols kt + hl*8 + 0..3
    float4 q1 = rowp[hl * 2 + 1];                  // cols kt + hl*8 + 4..7
    float4 q2 = rowp[4 + hl * 2];                  // cols kt + 16 + hl*8 + 0..3
    float4 q3 = rowp[4 + hl * 2 + 1];              // cols kt + 16 + hl*8 + 4..7
    float rv[16] = {q0.x, q0.y, q0.z, q0.w, q1.x, q1.y, q1.z, q1.w,
                    q2.x, q2.y, q2.z, q2.w, q3.x, q3.y, q3.z, q3.w};
    if (masked) {                                  // only the final tile
#pragma unroll
      for (int i = 0; i < 16; ++i) {
        int col = kt + ((i < 8) ? (hl * 8 + i) : (16 + hl * 8 + (i - 8)));
        if (col >= SEQ) rv[i] = NEGINF;
      }
    }
    // online softmax: 16-wide local reduce + single cross-half shuffle
    float pmax = rv[0];
#pragma unroll
    for (int i = 1; i < 16; ++i) pmax = fmaxf(pmax, rv[i]);
    pmax = fmaxf(pmax, __shfl_xor(pmax, 16));
    float nm = fmaxf(mrow, pmax);
    float fscale = __expf(mrow - nm);
    mrow = nm;
    float p[16], psum = 0.f;
#pragma unroll
    for (int i = 0; i < 16; ++i) { p[i] = __expf(rv[i] - nm); psum += p[i]; }
    psum += __shfl_xor(psum, 16);
    lrow = lrow * fscale + psum;
    // P fragment is already in A-layout: pack bf16 pairs
    FragPK pa;
#pragma unroll
    for (int j = 0; j < 8; ++j)
      pa.u[j] = (unsigned)f2bf(p[2 * j]) | ((unsigned)f2bf(p[2 * j + 1]) << 16);
    // rescale accumulators: broadcast per-row factor into D-layout lanes
#pragma unroll
    for (int r = 0; r < 8; ++r) {
      float fr = __shfl(fscale, r + hl * 8);
#pragma unroll
      for (int c = 0; c < 4; ++c) accd[c][r] *= fr;
    }
    // ---- O += P * V (4 d-chunks of 16) ----
#pragma unroll
    for (int c = 0; c < 4; ++c) {
      int d = c * 16 + ln;                         // lane = d column
      const unsigned short* Vrow = VT + ((size_t)bh * HDIM + d) * SEQP + kt;
      Frag32B bvf;
      bvf.q[0] = *(const uint4*)(Vrow + hl * 8);
      bvf.q[1] = *(const uint4*)(Vrow + 16 + hl * 8);
      accd[c] = __builtin_amdgcn_wmma_f32_16x16x32_bf16(false, pa.v, false, bvf.v,
                                                        (short)0, accd[c], false, false);
    }
    wave_lds_sync();                               // WAR vs next tile's stores
  };

  for (int kt = 0; kt < KT_FULL; kt += 32) kstep(kt, false);  // 46 unmasked tiles
  kstep(KT_FULL, true);                                       // masked tail tile

  if (qt < QTILES) {
    int b = bh / NHEAD, h = bh % NHEAD;
#pragma unroll
    for (int r = 0; r < 8; ++r) {
      float linv = 1.0f / __shfl(lrow, r + hl * 8);
      int tq = qt * 16 + r + hl * 8;
      if (tq < SEQ) {
        size_t base = (size_t)(b * SEQ + tq) * E_DIM + h * HDIM + ln;
#pragma unroll
        for (int c = 0; c < 4; ++c)
          AO[base + c * 16] = f2bf(accd[c][r] * linv);
      }
    }
  }
}

// ---------------- host-side orchestration ----------------
extern "C" void kernel_launch(void* const* d_in, const int* in_sizes, int n_in,
                              void* d_out, int out_size, void* d_ws, size_t ws_size,
                              hipStream_t stream) {
  (void)in_sizes; (void)n_in; (void)out_size; (void)ws_size;
  const float* hs  = (const float*)d_in[0];
  const float* q_w = (const float*)d_in[1];
  const float* q_b = (const float*)d_in[2];
  const float* k_w = (const float*)d_in[3];
  const float* v_w = (const float*)d_in[4];
  const float* v_b = (const float*)d_in[5];
  const float* o_w = (const float*)d_in[6];
  const float* o_b = (const float*)d_in[7];

  char* ws = (char*)d_ws;
  size_t off = 0;
  auto carve = [&](size_t bytes) -> void* {
    void* p = ws + off;
    off += (bytes + 255) & ~(size_t)255;
    return p;
  };
  const size_t QKV_ELEMS = (size_t)BATCH * NHEAD * SEQP * HDIM;  // 7,700,480
  unsigned short* Xbf = (unsigned short*)carve((size_t)MROWS * E_DIM * 2);
  unsigned short* Wqb = (unsigned short*)carve((size_t)E_DIM * E_DIM * 2);
  unsigned short* Wkb = (unsigned short*)carve((size_t)E_DIM * E_DIM * 2);
  unsigned short* Wvb = (unsigned short*)carve((size_t)E_DIM * E_DIM * 2);
  unsigned short* Wob = (unsigned short*)carve((size_t)E_DIM * E_DIM * 2);
  unsigned short* Qb  = (unsigned short*)carve(QKV_ELEMS * 2);   // contiguous…
  unsigned short* Kbf = (unsigned short*)carve(QKV_ELEMS * 2);   // …with Qb
  unsigned short* VTb = (unsigned short*)carve(QKV_ELEMS * 2);   // …and Kbf
  unsigned short* AO  = (unsigned short*)carve((size_t)MROWS * E_DIM * 2);

  // fp32 -> bf16 conversions
  int n = MROWS * E_DIM;
  f32_to_bf16_kernel<<<(n + 255) / 256, 256, 0, stream>>>(hs, Xbf, n);
  n = E_DIM * E_DIM;
  f32_to_bf16_kernel<<<(n + 255) / 256, 256, 0, stream>>>(q_w, Wqb, n);
  f32_to_bf16_kernel<<<(n + 255) / 256, 256, 0, stream>>>(k_w, Wkb, n);
  f32_to_bf16_kernel<<<(n + 255) / 256, 256, 0, stream>>>(v_w, Wvb, n);
  f32_to_bf16_kernel<<<(n + 255) / 256, 256, 0, stream>>>(o_w, Wob, n);

  // zero the seq-padded Q/K/V buffers in one pass (they are contiguous)
  zero_u16_kernel<<<2048, 256, 0, stream>>>(Qb, (int)(3 * QKV_ELEMS));

  // projections: 7500 16x64 wave-tiles, 8 waves/block -> 938 blocks
  const int gblocks = (GEMM_TILES + 7) / 8;
  gemm_wmma_kernel<0><<<gblocks, 256, 0, stream>>>(Xbf, Wqb, q_b, Qb);
  gemm_wmma_kernel<1><<<gblocks, 256, 0, stream>>>(Xbf, Wkb, nullptr, Kbf);
  gemm_wmma_kernel<2><<<gblocks, 256, 0, stream>>>(Xbf, Wvb, v_b, VTb);

  // flash attention: (B*H) * 24 blocks, 4 waves/block
  attn_flash_kernel<<<BATCH * NHEAD * BLOCKS_PER_BH, 128, 0, stream>>>(Qb, Kbf, VTb, AO);

  // output projection -> fp32 d_out
  gemm_wmma_kernel<3><<<gblocks, 256, 0, stream>>>(AO, Wob, o_b, d_out);
}
